// Decoder_Attn_44014824849456
// MI455X (gfx1250) — matrix operand order repaired and checked by reading, not compiled, not run on hardware
//
#include <hip/hip_runtime.h>
#include <hip/hip_bf16.h>
#include <math.h>

#define S_   512
#define B_   64
#define H_   1024
#define E_   512
#define V_   32000
#define TWOH 2048
#define DIN  2560   // 2H + E
#define G4   4096   // 4H
#define NCH  8      // S chunks for online softmax
#define SCH  64     // S per chunk

typedef __attribute__((ext_vector_type(2)))  float  v2f;
typedef __attribute__((ext_vector_type(4)))  float  v4f;
typedef __attribute__((ext_vector_type(8)))  float  v8f;
typedef __attribute__((ext_vector_type(16))) __bf16 v16bf;

#if __has_builtin(__builtin_amdgcn_wmma_f32_16x16x4_f32)
#define HAVE_WMMA_F32X4 1
#endif

__device__ __forceinline__ float warp_sum(float v) {
#pragma unroll
  for (int off = 16; off > 0; off >>= 1)
    v += __shfl_down(v, off, 32);
  return v;
}

// ---------------------------------------------------------------------------
// Kernel 1: fused attention (online softmax over an S-chunk) producing
// partial context vectors + (m, l) per (batch, chunk). Reads ES exactly once.
// ---------------------------------------------------------------------------
__global__ __launch_bounds__(256)
void attn_partial_kernel(const float* __restrict__ ES,
                         const float* __restrict__ h0,
                         const float* __restrict__ W_energy,
                         const float* __restrict__ b_energy,
                         float* __restrict__ part_ctx,
                         float* __restrict__ part_ml) {
  const int b    = blockIdx.x;
  const int ch   = blockIdx.y;
  const int tid  = threadIdx.x;
  const int lane = tid & 31;
  const int wid  = tid >> 5;
  __shared__ float red[8];

  // hdot = h0[b] . W_energy[0:H] + b_energy  (recomputed per block; 4 KB read)
  const float* h0row = h0 + (size_t)b * H_;
  float hp = 0.f;
  for (int k = tid; k < H_; k += 256) hp += h0row[k] * W_energy[k];
  hp = warp_sum(hp);
  if (lane == 0) red[wid] = hp;
  __syncthreads();
  const float hdot = red[0] + red[1] + red[2] + red[3] +
                     red[4] + red[5] + red[6] + red[7] + b_energy[0];
  __syncthreads();

  // each thread owns 8 of the 2048 channels: d0 = 4*tid, d1 = 1024 + 4*tid
  const float* we = W_energy + H_;
  const int d0 = 4 * tid;
  const int d1 = 1024 + 4 * tid;
  const v4f w0 = *(const v4f*)(we + d0);
  const v4f w1 = *(const v4f*)(we + d1);

  v4f c0v = {0.f, 0.f, 0.f, 0.f};
  v4f c1v = {0.f, 0.f, 0.f, 0.f};
  float m = -1e30f, l = 0.f;

  const int s0 = ch * SCH;
  for (int s = s0; s < s0 + SCH; ++s) {
    const float* esr = ES + ((size_t)s * B_ + b) * TWOH;
    const v4f e0 = *(const v4f*)(esr + d0);
    const v4f e1 = *(const v4f*)(esr + d1);
    float dp = e0.x * w0.x + e0.y * w0.y + e0.z * w0.z + e0.w * w0.w +
               e1.x * w1.x + e1.y * w1.y + e1.z * w1.z + e1.w * w1.w;
    dp = warp_sum(dp);
    if (lane == 0) red[wid] = dp;
    __syncthreads();
    float ener = red[0] + red[1] + red[2] + red[3] +
                 red[4] + red[5] + red[6] + red[7] + hdot;
    __syncthreads();
    ener = fmaxf(ener, 0.f);                 // relu
    const float mn = fmaxf(m, ener);
    const float sc = __expf(m - mn);
    const float p  = __expf(ener - mn);
    l = l * sc + p;
    c0v = c0v * sc + e0 * p;
    c1v = c1v * sc + e1 * p;
    m = mn;
  }

  float* dst = part_ctx + (size_t)(b * NCH + ch) * TWOH;
  *(v4f*)(dst + d0) = c0v;
  *(v4f*)(dst + d1) = c1v;
  if (tid == 0) {
    part_ml[(b * NCH + ch) * 2 + 0] = m;
    part_ml[(b * NCH + ch) * 2 + 1] = l;
  }
}

// ---------------------------------------------------------------------------
// Kernel 2: combine chunk partials -> context, and build rnn_in = [ctx, emb]
// ---------------------------------------------------------------------------
__global__ __launch_bounds__(256)
void attn_combine_kernel(const float* __restrict__ part_ctx,
                         const float* __restrict__ part_ml,
                         const int* __restrict__ x,
                         const float* __restrict__ emb_table,
                         float* __restrict__ rnn_in) {
  const int b = blockIdx.x, tid = threadIdx.x;
  float mv[NCH], lv[NCH], w[NCH];
  float M = -1e30f;
#pragma unroll
  for (int i = 0; i < NCH; ++i) {
    mv[i] = part_ml[(b * NCH + i) * 2 + 0];
    lv[i] = part_ml[(b * NCH + i) * 2 + 1];
    M = fmaxf(M, mv[i]);
  }
  float L = 0.f;
#pragma unroll
  for (int i = 0; i < NCH; ++i) { w[i] = __expf(mv[i] - M); L += lv[i] * w[i]; }
  const float invL = 1.0f / L;

  const int d0 = 4 * tid;
  const int d1 = 1024 + 4 * tid;
  v4f a0 = {0.f, 0.f, 0.f, 0.f};
  v4f a1 = {0.f, 0.f, 0.f, 0.f};
#pragma unroll
  for (int i = 0; i < NCH; ++i) {
    const float* src = part_ctx + (size_t)(b * NCH + i) * TWOH;
    a0 += *(const v4f*)(src + d0) * w[i];
    a1 += *(const v4f*)(src + d1) * w[i];
  }
  float* out = rnn_in + (size_t)b * DIN;
  *(v4f*)(out + d0) = a0 * invL;
  *(v4f*)(out + d1) = a1 * invL;

  if (tid < E_ / 4) {                      // embedding gather (512 floats)
    const int row = x[b];
    *(v4f*)(out + TWOH + 4 * tid) = *(const v4f*)(emb_table + (size_t)row * E_ + 4 * tid);
  }
}

// ---------------------------------------------------------------------------
// WMMA 16x16 tile: acc += A[16xK] * W^T  where B[k][n] = W[n][k] (row-major W)
// Fragment layouts per CDNA5 ISA 7.12.2.
// ---------------------------------------------------------------------------
__device__ __forceinline__
v8f wmma_tile(const float* __restrict__ A, int lda,
              const float* __restrict__ W, int ldw,
              int K, v8f acc) {
  const int lane = threadIdx.x & 31;
  const int lo = lane & 15;
  const int hi = lane >> 4;
  const float* aRow = A + (size_t)lo * lda;
  const float* wRow = W + (size_t)lo * ldw;
#ifdef HAVE_WMMA_F32X4
  for (int k = 0; k < K; k += 4) {
    const v2f a = *(const v2f*)(aRow + k + 2 * hi);   // A[lo][k + 2*hi + {0,1}]
    const v2f b = *(const v2f*)(wRow + k + 2 * hi);   // B[k + 2*hi + {0,1}][lo]
    acc = __builtin_amdgcn_wmma_f32_16x16x4_f32(false, a, false, b,
                                                (short)0, acc, false, false);
  }
#else
  // fallback: codegen-confirmed bf16 WMMA, converting f32 operands on the fly
  for (int k = 0; k < K; k += 32) {
    v16bf a, b;
#pragma unroll
    for (int e = 0; e < 16; ++e) {
      a[e] = (__bf16)aRow[k + ((e < 8) ? 0 : 16) + 8 * hi + (e & 7)];
      b[e] = (__bf16)wRow[k + ((e < 8) ? 0 : 16) + 8 * hi + (e & 7)];
    }
    acc = __builtin_amdgcn_wmma_f32_16x16x32_bf16(false, a, false, b,
                                                  (short)0, acc, false, false);
  }
#endif
  return acc;
}

// ---------------------------------------------------------------------------
// Kernel 3: gates = rnn_in @ W_ih^T + h0 @ W_hh^T + b_ih + b_hh
// one wave per 16x16 tile: 4 M-tiles x 256 N-tiles = 1024 waves
// ---------------------------------------------------------------------------
__global__ __launch_bounds__(256)
void gates_kernel(const float* __restrict__ rnn_in,
                  const float* __restrict__ h0,
                  const float* __restrict__ W_ih,
                  const float* __restrict__ W_hh,
                  const float* __restrict__ b_ih,
                  const float* __restrict__ b_hh,
                  float* __restrict__ gates) {
  const int wave  = threadIdx.x >> 5;
  const int tile  = blockIdx.x * 8 + wave;
  const int mBase = (tile & 3) * 16;
  const int nBase = (tile >> 2) * 16;
  const int lane  = threadIdx.x & 31;
  const int lo = lane & 15, hi = lane >> 4;
  const int n = nBase + lo;

  const float bias = b_ih[n] + b_hh[n];    // bias depends only on column N
  v8f acc;
#pragma unroll
  for (int v = 0; v < 8; ++v) acc[v] = bias;

  acc = wmma_tile(rnn_in + (size_t)mBase * DIN, DIN,
                  W_ih   + (size_t)nBase * DIN, DIN, DIN, acc);
  acc = wmma_tile(h0   + (size_t)mBase * H_, H_,
                  W_hh + (size_t)nBase * H_, H_, H_, acc);

#pragma unroll
  for (int v = 0; v < 8; ++v)              // D row M = v + 8*hi, col N = lo
    gates[(size_t)(mBase + 8 * hi + v) * G4 + n] = acc[v];
}

// ---------------------------------------------------------------------------
// Kernel 4: LSTM cell elementwise
// ---------------------------------------------------------------------------
__global__ __launch_bounds__(256)
void lstm_kernel(const float* __restrict__ gates,
                 const float* __restrict__ c0,
                 float* __restrict__ h1,
                 float* __restrict__ c1) {
  const int idx = blockIdx.x * 256 + threadIdx.x;
  if (idx >= B_ * H_) return;
  const int b = idx >> 10;
  const int h = idx & 1023;
  const float* g = gates + (size_t)b * G4;
  const float gi = g[h], gf = g[H_ + h], gg = g[2 * H_ + h], go = g[3 * H_ + h];
  const float si = 1.f / (1.f + __expf(-gi));
  const float sf = 1.f / (1.f + __expf(-gf));
  const float so = 1.f / (1.f + __expf(-go));
  const float c = sf * c0[idx] + si * tanhf(gg);
  c1[idx] = c;
  h1[idx] = so * tanhf(c);
}

// ---------------------------------------------------------------------------
// Kernel 5: pred = h1 @ W_fc^T + b_fc   (M=64, N=32000, K=1024)
// 4 x 2000 = 8000 tiles, one wave each
// ---------------------------------------------------------------------------
__global__ __launch_bounds__(256)
void pred_kernel(const float* __restrict__ h1,
                 const float* __restrict__ W_fc,
                 const float* __restrict__ b_fc,
                 float* __restrict__ pred) {
  const int wave  = threadIdx.x >> 5;
  const int tile  = blockIdx.x * 8 + wave;
  const int mBase = (tile & 3) * 16;
  const int nBase = (tile >> 2) * 16;
  const int lane  = threadIdx.x & 31;
  const int lo = lane & 15, hi = lane >> 4;
  const int n = nBase + lo;

  const float bias = b_fc[n];
  v8f acc;
#pragma unroll
  for (int v = 0; v < 8; ++v) acc[v] = bias;

  acc = wmma_tile(h1   + (size_t)mBase * H_, H_,
                  W_fc + (size_t)nBase * H_, H_, H_, acc);

#pragma unroll
  for (int v = 0; v < 8; ++v)
    pred[(size_t)(mBase + 8 * hi + v) * V_ + n] = acc[v];
}

// ---------------------------------------------------------------------------
extern "C" void kernel_launch(void* const* d_in, const int* in_sizes, int n_in,
                              void* d_out, int out_size, void* d_ws, size_t ws_size,
                              hipStream_t stream) {
  (void)in_sizes; (void)n_in; (void)out_size; (void)ws_size;
  const int*   x        = (const int*)  d_in[0];
  const float* ES       = (const float*)d_in[1];
  const float* hidden   = (const float*)d_in[2];   // h0 (1,B,H)
  const float* cell     = (const float*)d_in[3];   // c0 (1,B,H)
  const float* emb      = (const float*)d_in[4];
  const float* W_energy = (const float*)d_in[5];
  const float* b_energy = (const float*)d_in[6];
  const float* W_ih     = (const float*)d_in[7];
  const float* W_hh     = (const float*)d_in[8];
  const float* b_ih     = (const float*)d_in[9];
  const float* b_hh     = (const float*)d_in[10];
  const float* W_fc     = (const float*)d_in[11];
  const float* b_fc     = (const float*)d_in[12];

  float* out  = (float*)d_out;
  float* pred = out;                               // (B, V)
  float* h1   = out + (size_t)B_ * V_;             // (1, B, H)
  float* c1   = h1 + (size_t)B_ * H_;              // (1, B, H)

  float* ws       = (float*)d_ws;
  float* part_ctx = ws;                                   // B*NCH*2048
  float* part_ml  = part_ctx + (size_t)B_ * NCH * TWOH;   // B*NCH*2
  float* rnn_in   = part_ml + (size_t)B_ * NCH * 2;       // B*2560
  float* gates    = rnn_in + (size_t)B_ * DIN;            // B*4096

  attn_partial_kernel<<<dim3(B_, NCH), 256, 0, stream>>>(
      ES, hidden, W_energy, b_energy, part_ctx, part_ml);
  attn_combine_kernel<<<dim3(B_), 256, 0, stream>>>(
      part_ctx, part_ml, x, emb, rnn_in);
  gates_kernel<<<dim3(128), 256, 0, stream>>>(
      rnn_in, hidden, W_ih, W_hh, b_ih, b_hh, gates);
  lstm_kernel<<<dim3((B_ * H_ + 255) / 256), 256, 0, stream>>>(
      gates, cell, h1, c1);
  pred_kernel<<<dim3(1000), 256, 0, stream>>>(
      h1, W_fc, b_fc, pred);
}
